// CustomModel_4028679324234
// MI455X (gfx1250) — matrix operand order repaired and checked by reading, not compiled, not run on hardware
//
#include <hip/hip_runtime.h>
#include <cmath>

#define HEADS 4
#define EDIM 8
#define NEG_SLOPE 0.2f
#define GAT_EPS 1e-16f

typedef __attribute__((ext_vector_type(16))) _Float16 v16h;
typedef __attribute__((ext_vector_type(8)))  float    v8f;

// ---------------------------------------------------------------- utilities
__global__ void fill_kernel(float* __restrict__ p, long n, float v) {
    long i = blockIdx.x * (long)blockDim.x + threadIdx.x;
    if (i < n) p[i] = v;
}

// q[ed*H + h] = sum_c We[ed*(H*C)+h*C+c] * a_edge[h*C+c]
template<int C>
__global__ void q_kernel(const float* __restrict__ We,
                         const float* __restrict__ a_edge,
                         float* __restrict__ q) {
    int t = threadIdx.x;
    if (t >= EDIM * HEADS) return;
    int ed = t / HEADS, h = t % HEADS;
    float s = 0.f;
    for (int c = 0; c < C; ++c)
        s += We[ed * (HEADS * C) + h * C + c] * a_edge[h * C + c];
    q[ed * HEADS + h] = s;
}

// ------------------------------------------------------------ WMMA node GEMM
// Out[nrows x M] = A[nrows x K] (f32 -> f16) @ W[K x M] (f32 -> f16, LDS)
// Block: 128 threads = 4 waves; 16 output rows per block; wave owns TPW
// 16-wide column tiles. W is staged into LDS *pre-swizzled in WMMA B-fragment
// order*, so each lane's 16 B halves are one contiguous 32-byte LDS read.
template<int K, int M, int TPW>
__global__ void gemm_f16_wmma_kernel(const float* __restrict__ A,
                                     const float* __restrict__ W,
                                     float* __restrict__ Out, int nrows) {
    constexpr int KSTEPS = K / 32;
    constexpr int CTILES = M / 16;
    __shared__ _Float16 sW[K * M];   // [ks][ct][lane][16 halves]

    const int tid = threadIdx.x;
    // swizzled fill: coalesced global f32 reads, scattered ds_store_b16
    for (int idx = tid; idx < K * M; idx += blockDim.x) {
        const int k  = idx / M;
        const int m  = idx - k * M;
        const int ks = k >> 5;            // k-step of 32
        const int kr = k & 31;
        const int fh = (kr >> 3) & 1;     // which lane-half supplies this k
        const int fi = (kr & 7) | ((kr >> 4) << 3); // element within fragment
        const int ct = m >> 4;            // column tile
        const int cl = m & 15;            // column within tile
        const int lane16 = fh * 16 + cl;  // owning lane
        sW[(((ks * CTILES + ct) * 32) + lane16) * 16 + fi] = (_Float16)W[idx];
    }
    __syncthreads();

    const int lane = tid & 31;
    const int wave = tid >> 5;
    const int r0   = blockIdx.x * 16;
    const int grow = r0 + (lane & 15);
    const int hi   = lane >> 4;              // 0: lanes 0-15, 1: lanes 16-31
    const bool fullblk = (r0 + 16 <= nrows);

    // A fragments (ISA 16-bit A layout): lane holds row (lane&15);
    // halves 0..7 -> K = kb + hi*8 + i, halves 8..15 -> K = kb+16+hi*8+(i-8)
    v16h afrag[KSTEPS];
    const float* arow = A + (long)grow * K + hi * 8;
    for (int ksi = 0; ksi < KSTEPS; ++ksi) {
        float4 x0 = make_float4(0.f, 0.f, 0.f, 0.f), x1 = x0, y0 = x0, y1 = x0;
        if (fullblk || grow < nrows) {
            const float* ap = arow + ksi * 32;
            x0 = ((const float4*)ap)[0];
            x1 = ((const float4*)ap)[1];
            y0 = ((const float4*)(ap + 16))[0];
            y1 = ((const float4*)(ap + 16))[1];
        }
        v16h a;
        a[0]  = (_Float16)x0.x; a[1]  = (_Float16)x0.y;
        a[2]  = (_Float16)x0.z; a[3]  = (_Float16)x0.w;
        a[4]  = (_Float16)x1.x; a[5]  = (_Float16)x1.y;
        a[6]  = (_Float16)x1.z; a[7]  = (_Float16)x1.w;
        a[8]  = (_Float16)y0.x; a[9]  = (_Float16)y0.y;
        a[10] = (_Float16)y0.z; a[11] = (_Float16)y0.w;
        a[12] = (_Float16)y1.x; a[13] = (_Float16)y1.y;
        a[14] = (_Float16)y1.z; a[15] = (_Float16)y1.w;
        afrag[ksi] = a;
    }

    for (int t = 0; t < TPW; ++t) {
        const int ct  = wave * TPW + t;
        const int col = ct * 16 + (lane & 15);
        v8f c = {};
        for (int ksi = 0; ksi < KSTEPS; ++ksi) {
            const v16h b = *(const v16h*)
                &sW[(((ksi * CTILES + ct) * 32) + lane) * 16];
            c = __builtin_amdgcn_wmma_f32_16x16x32_f16(
                    false, afrag[ksi], false, b, (short)0, c, false, false);
        }
        // C/D layout: VGPR r -> row (r0 + hi*8 + r), col = lane&15
        float* outp = Out + (long)(r0 + hi * 8) * M + col;
        if (fullblk) {
            for (int r = 0; r < 8; ++r) outp[r * M] = c[r];
        } else {
            for (int r = 0; r < 8; ++r)
                if (r0 + hi * 8 + r < nrows) outp[r * M] = c[r];
        }
    }
}

// ---------------------------------------------------- per-(node,head) coeffs
template<int C>
__global__ void attn_kernel(const float* __restrict__ h,
                            const float* __restrict__ a_src,
                            const float* __restrict__ a_dst,
                            float* __restrict__ as_, float* __restrict__ ad_,
                            long nnodes) {
    long i = blockIdx.x * (long)blockDim.x + threadIdx.x;
    if (i >= nnodes * HEADS) return;
    const int hh = (int)(i & (HEADS - 1));
    const long n = i >> 2;
    const float* hp = h + (n * HEADS + hh) * C;
    float s = 0.f, d = 0.f;
    for (int c = 0; c < C; ++c) {
        float v = hp[c];
        s += v * a_src[hh * C + c];
        d += v * a_dst[hh * C + c];
    }
    as_[n * HEADS + hh] = s;
    ad_[n * HEADS + hh] = d;
}

// ----------------------------------------------------------- edge primitives
__device__ __forceinline__ float edge_alpha(long e, int hh, long Eedges,
                                            const long long* __restrict__ ei,
                                            const float* __restrict__ ea,
                                            const float* __restrict__ q,
                                            const float* __restrict__ as_,
                                            const float* __restrict__ ad_,
                                            long& src, long& dst) {
    float ae = 0.f;
    if (e < Eedges) {
        src = (long)ei[e];
        dst = (long)ei[Eedges + e];
        const float* eap = ea + e * EDIM;
        for (int d = 0; d < EDIM; ++d) ae += eap[d] * q[d * HEADS + hh];
    } else {                       // self loop, zero edge features
        src = dst = e - Eedges;
    }
    float al = as_[src * HEADS + hh] + ad_[dst * HEADS + hh] + ae;
    return al > 0.f ? al : al * NEG_SLOPE;     // leaky_relu
}

__global__ void edge_max_kernel(const long long* __restrict__ ei,
                                const float* __restrict__ ea,
                                const float* __restrict__ q,
                                const float* __restrict__ as_,
                                const float* __restrict__ ad_,
                                float* __restrict__ amax,
                                long Eedges, long Etot) {
    long i = blockIdx.x * (long)blockDim.x + threadIdx.x;
    if (i >= Etot * HEADS) return;
    const int hh = (int)(i & (HEADS - 1));
    const long e = i >> 2;
    long src, dst;
    float al = edge_alpha(e, hh, Eedges, ei, ea, q, as_, ad_, src, dst);
    __hip_atomic_fetch_max(&amax[dst * HEADS + hh], al,
                           __ATOMIC_RELAXED, __HIP_MEMORY_SCOPE_AGENT);
}

__global__ void edge_sum_kernel(const long long* __restrict__ ei,
                                const float* __restrict__ ea,
                                const float* __restrict__ q,
                                const float* __restrict__ as_,
                                const float* __restrict__ ad_,
                                const float* __restrict__ amax,
                                float* __restrict__ den,
                                long Eedges, long Etot) {
    long i = blockIdx.x * (long)blockDim.x + threadIdx.x;
    if (i >= Etot * HEADS) return;
    const int hh = (int)(i & (HEADS - 1));
    const long e = i >> 2;
    long src, dst;
    float al = edge_alpha(e, hh, Eedges, ei, ea, q, as_, ad_, src, dst);
    float ex = __expf(al - amax[dst * HEADS + hh]);
    unsafeAtomicAdd(&den[dst * HEADS + hh], ex);
}

template<int C>
__global__ void edge_agg_kernel(const long long* __restrict__ ei,
                                const float* __restrict__ ea,
                                const float* __restrict__ q,
                                const float* __restrict__ as_,
                                const float* __restrict__ ad_,
                                const float* __restrict__ amax,
                                const float* __restrict__ den,
                                const float* __restrict__ h,
                                float* __restrict__ acc,
                                long Eedges, long Etot) {
    long i = blockIdx.x * (long)blockDim.x + threadIdx.x;
    if (i >= Etot * HEADS) return;
    const int hh = (int)(i & (HEADS - 1));
    const long e = i >> 2;
    long src, dst;
    float al = edge_alpha(e, hh, Eedges, ei, ea, q, as_, ad_, src, dst);
    float ex = __expf(al - amax[dst * HEADS + hh]);
    float w  = ex / (den[dst * HEADS + hh] + GAT_EPS);
    const float* hs = h + (src * HEADS + hh) * C;
    float* op = acc + (dst * HEADS + hh) * C;
    for (int c = 0; c < C; ++c) unsafeAtomicAdd(&op[c], w * hs[c]);
}

// head-mean + bias + ELU
template<int C>
__global__ void mean_bias_elu_kernel(const float* __restrict__ acc,
                                     const float* __restrict__ b,
                                     float* __restrict__ out, long nnodes) {
    long i = blockIdx.x * (long)blockDim.x + threadIdx.x;
    if (i >= nnodes * C) return;
    const int  c = (int)(i % C);
    const long n = i / C;
    float s = 0.f;
    for (int h = 0; h < HEADS; ++h) s += acc[(n * HEADS + h) * C + c];
    s = s * (1.0f / HEADS) + b[c];
    out[n * C + c] = s > 0.f ? s : expm1f(s);
}

// ------------------------------------------------------------------ launcher
extern "C" void kernel_launch(void* const* d_in, const int* in_sizes, int n_in,
                              void* d_out, int out_size, void* d_ws, size_t ws_size,
                              hipStream_t stream) {
    (void)n_in; (void)out_size; (void)ws_size;
    const float*     x       = (const float*)d_in[0];
    const long long* ei      = (const long long*)d_in[1];
    const float*     ea      = (const float*)d_in[2];
    const float*     W1      = (const float*)d_in[3];
    const float*     We1     = (const float*)d_in[4];
    const float*     a_src1  = (const float*)d_in[5];
    const float*     a_dst1  = (const float*)d_in[6];
    const float*     a_edge1 = (const float*)d_in[7];
    const float*     b1      = (const float*)d_in[8];
    const float*     W2      = (const float*)d_in[9];
    const float*     We2     = (const float*)d_in[10];
    const float*     a_src2  = (const float*)d_in[11];
    const float*     a_dst2  = (const float*)d_in[12];
    const float*     a_edge2 = (const float*)d_in[13];
    const float*     b2      = (const float*)d_in[14];

    constexpr int FIN = 64, C1 = 32, C2 = 16;
    const long nN   = in_sizes[0] / FIN;
    const long nE   = in_sizes[1] / 2;
    const long Etot = nE + nN;

    // workspace layout (floats); h/acc regions reused in layer 2
    float* f    = (float*)d_ws;
    float* h1   = f;                      // N*128 (layer2 h2 uses first N*64)
    float* acc  = h1   + nN * 128;        // N*128 (layer2 acc uses first N*64)
    float* hin2 = acc  + nN * 128;        // N*32
    float* as_  = hin2 + nN * 32;         // N*H
    float* ad_  = as_  + nN * HEADS;      // N*H
    float* amax = ad_  + nN * HEADS;      // N*H
    float* den  = amax + nN * HEADS;      // N*H
    float* q    = den  + nN * HEADS;      // EDIM*H = 32

    auto cdiv = [](long a, long b) { return (int)((a + b - 1) / b); };
    const int TPB = 256;
    const float NEGINF = -INFINITY;
    const int eg = cdiv(Etot * HEADS, TPB);
    const int nh = cdiv(nN * HEADS, TPB);

    // ----------------------------- layer 1 -----------------------------
    fill_kernel<<<cdiv(nN * 128, TPB), TPB, 0, stream>>>(acc, nN * 128, 0.f);
    fill_kernel<<<nh, TPB, 0, stream>>>(amax, nN * HEADS, NEGINF);
    fill_kernel<<<nh, TPB, 0, stream>>>(den, nN * HEADS, 0.f);
    q_kernel<C1><<<1, 32, 0, stream>>>(We1, a_edge1, q);
    gemm_f16_wmma_kernel<64, 128, 2><<<cdiv(nN, 16), 128, 0, stream>>>(x, W1, h1, (int)nN);
    attn_kernel<C1><<<nh, TPB, 0, stream>>>(h1, a_src1, a_dst1, as_, ad_, nN);
    edge_max_kernel<<<eg, TPB, 0, stream>>>(ei, ea, q, as_, ad_, amax, nE, Etot);
    edge_sum_kernel<<<eg, TPB, 0, stream>>>(ei, ea, q, as_, ad_, amax, den, nE, Etot);
    edge_agg_kernel<C1><<<eg, TPB, 0, stream>>>(ei, ea, q, as_, ad_, amax, den, h1, acc, nE, Etot);
    mean_bias_elu_kernel<C1><<<cdiv(nN * C1, TPB), TPB, 0, stream>>>(acc, b1, hin2, nN);

    // ----------------------------- layer 2 -----------------------------
    fill_kernel<<<cdiv(nN * 64, TPB), TPB, 0, stream>>>(acc, nN * 64, 0.f);
    fill_kernel<<<nh, TPB, 0, stream>>>(amax, nN * HEADS, NEGINF);
    fill_kernel<<<nh, TPB, 0, stream>>>(den, nN * HEADS, 0.f);
    q_kernel<C2><<<1, 32, 0, stream>>>(We2, a_edge2, q);
    gemm_f16_wmma_kernel<32, 64, 1><<<cdiv(nN, 16), 128, 0, stream>>>(hin2, W2, h1, (int)nN);
    attn_kernel<C2><<<nh, TPB, 0, stream>>>(h1, a_src2, a_dst2, as_, ad_, nN);
    edge_max_kernel<<<eg, TPB, 0, stream>>>(ei, ea, q, as_, ad_, amax, nE, Etot);
    edge_sum_kernel<<<eg, TPB, 0, stream>>>(ei, ea, q, as_, ad_, amax, den, nE, Etot);
    edge_agg_kernel<C2><<<eg, TPB, 0, stream>>>(ei, ea, q, as_, ad_, amax, den, h1, acc, nE, Etot);
    mean_bias_elu_kernel<C2><<<cdiv(nN * C2, TPB), TPB, 0, stream>>>(acc, b2, (float*)d_out, nN);
}
